// HungarianMatcher_6047313953204
// MI455X (gfx1250) — matrix operand order repaired and checked by reading, not compiled, not run on hardware
//
#include <hip/hip_runtime.h>
#include <hip/hip_bf16.h>

typedef __attribute__((ext_vector_type(16))) _Float16 v16h;
typedef __attribute__((ext_vector_type(8)))  _Float16 v8h;
typedef __attribute__((ext_vector_type(4)))  _Float16 v4h;
typedef __attribute__((ext_vector_type(8)))  float    v8f;

#define BN_ROWS 1600   // B*N
#define HW      16384  // 128*128
#define T_ROWS  320
#define NCLS    80
#define SQ      100    // hungarian square size
#define PERIMG  20
#define NB      16

__device__ __forceinline__ float sigmoidf(float x) { return 1.0f / (1.0f + __expf(-x)); }

// ---------------- pass 1: pm = sigmoid(pred_masks) -> f16, row sum of pm^2 ----
__global__ void __launch_bounds__(256) pm_sigmoid_kernel(
    const float* __restrict__ in, _Float16* __restrict__ pm, float* __restrict__ pmsq) {
  __shared__ float red[256];
  const int row = blockIdx.x;
  const float4* rin = (const float4*)(in + (size_t)row * HW);
  v4h* rout = (v4h*)(pm + (size_t)row * HW);
  float acc = 0.f;
  #pragma unroll
  for (int it = 0; it < HW / 1024; ++it) {
    int idx = it * 256 + threadIdx.x;
    float4 x = rin[idx];
    float s0 = sigmoidf(x.x), s1 = sigmoidf(x.y), s2 = sigmoidf(x.z), s3 = sigmoidf(x.w);
    acc += s0 * s0 + s1 * s1 + s2 * s2 + s3 * s3;
    v4h o = {(_Float16)s0, (_Float16)s1, (_Float16)s2, (_Float16)s3};
    rout[idx] = o;
  }
  red[threadIdx.x] = acc;
  __syncthreads();
  for (int o = 128; o > 0; o >>= 1) {
    if (threadIdx.x < o) red[threadIdx.x] += red[threadIdx.x + o];
    __syncthreads();
  }
  if (threadIdx.x == 0) pmsq[row] = red[0];
}

// ---------------- pass 2: tm = 2x2-average downsample(tgt_masks) -> f16, row sum tm^2
__global__ void __launch_bounds__(256) tm_resize_kernel(
    const float* __restrict__ in, _Float16* __restrict__ tm, float* __restrict__ tmsq) {
  __shared__ float red[256];
  const int t = blockIdx.x;
  const float* rin = in + (size_t)t * 256 * 256;
  _Float16* rout = tm + (size_t)t * HW;
  float acc = 0.f;
  for (int it = 0; it < HW / 256; ++it) {
    int o = it * 256 + threadIdx.x;
    int i = o >> 7, j = o & 127;
    float2 a = *(const float2*)(rin + (size_t)(2 * i) * 256 + 2 * j);
    float2 b = *(const float2*)(rin + (size_t)(2 * i + 1) * 256 + 2 * j);
    float v = 0.25f * (a.x + a.y + b.x + b.y);  // half-pixel 2x downsample == 2x2 mean
    acc += v * v;
    rout[o] = (_Float16)v;
  }
  red[threadIdx.x] = acc;
  __syncthreads();
  for (int o = 128; o > 0; o >>= 1) {
    if (threadIdx.x < o) red[threadIdx.x] += red[threadIdx.x + o];
    __syncthreads();
  }
  if (threadIdx.x == 0) tmsq[t] = red[0];
}

// ---------------- pass 3: WMMA GEMM (f16 in, f32 acc) + fused dice/prob epilogue
// M=1600, N=320, K=16384. 32x32 output per wave (2x2 register tile of 16x16 WMMA
// tiles): 2 VMEM b128 issues per v_wmma, 4 independent back-to-back WMMAs per
// k-step. 500 wave-tiles -> 125 blocks x 4 waves, n-major so a block shares B.
__global__ void __launch_bounds__(128) gemm_dice_kernel(
    const _Float16* __restrict__ pm, const _Float16* __restrict__ tm,
    const float* __restrict__ pmsq, const float* __restrict__ tmsq,
    const float* __restrict__ logits, const int* __restrict__ tids,
    float* __restrict__ Cout) {
  const int wave = threadIdx.x >> 5;
  const int lane = threadIdx.x & 31;
  const int tile = blockIdx.x * 4 + wave;  // 500 tiles of 32x32
  const int nt2 = tile / 50;               // 0..9
  const int mt2 = tile % 50;               // 0..49
  const int hi = lane >> 4;                // 0 or 1 (upper half-wave)
  const int ln = lane & 15;
  // A fragment: lane holds row; 16 contiguous halves at K+16*hi
  const _Float16* a0 = pm + (size_t)(mt2 * 32 + ln) * HW + (hi << 4);
  const _Float16* a1 = a0 + (size_t)16 * HW;
  // B fragment: lane holds tm row; 16 contiguous halves at K+16*hi
  const _Float16* b0 = tm + (size_t)(nt2 * 32 + ln) * HW + (hi << 4);
  const _Float16* b1 = b0 + (size_t)16 * HW;
  v8f acc00 = {}, acc01 = {}, acc10 = {}, acc11 = {};
  #pragma unroll 2
  for (int kk = 0; kk < HW; kk += 32) {
    union { v16h v; v8h h[2]; } ua0, ua1;
    ua0.h[0] = *(const v8h*)(a0 + kk);
    ua0.h[1] = *(const v8h*)(a0 + kk + 8);
    ua1.h[0] = *(const v8h*)(a1 + kk);
    ua1.h[1] = *(const v8h*)(a1 + kk + 8);
    v16h vb0 = *(const v16h*)(b0 + kk);
    v16h vb1 = *(const v16h*)(b1 + kk);
    acc00 = __builtin_amdgcn_wmma_f32_16x16x32_f16(false, ua0.v, false, vb0,
                                                   (short)0, acc00, false, false);
    acc01 = __builtin_amdgcn_wmma_f32_16x16x32_f16(false, ua0.v, false, vb1,
                                                   (short)0, acc01, false, false);
    acc10 = __builtin_amdgcn_wmma_f32_16x16x32_f16(false, ua1.v, false, vb0,
                                                   (short)0, acc10, false, false);
    acc11 = __builtin_amdgcn_wmma_f32_16x16x32_f16(false, ua1.v, false, vb1,
                                                   (short)0, acc11, false, false);
  }
  // epilogue: D element (VGPR r, lane) -> m = r + 8*hi, n = ln within each 16x16 tile
  const int t0 = nt2 * 32 + ln;
  const int t1 = t0 + 16;
  const float tden0 = tmsq[t0];
  const float tden1 = tmsq[t1];
  const int cls0 = tids[t0];
  const int cls1 = tids[t1];
  #pragma unroll
  for (int r = 0; r < 8; ++r) {
    const int q0 = mt2 * 32 + r + (hi << 3);
    const int q1 = q0 + 16;
    const float den_q0 = pmsq[q0], den_q1 = pmsq[q1];
    float d00 = (2.0f * acc00[r]) / (den_q0 + tden0 + 1e-4f);
    float d01 = (2.0f * acc01[r]) / (den_q0 + tden1 + 1e-4f);
    float d10 = (2.0f * acc10[r]) / (den_q1 + tden0 + 1e-4f);
    float d11 = (2.0f * acc11[r]) / (den_q1 + tden1 + 1e-4f);
    float p00 = sigmoidf(logits[q0 * NCLS + cls0]);
    float p01 = sigmoidf(logits[q0 * NCLS + cls1]);
    float p10 = sigmoidf(logits[q1 * NCLS + cls0]);
    float p11 = sigmoidf(logits[q1 * NCLS + cls1]);
    Cout[(size_t)q0 * T_ROWS + t0] = powf(d00, 0.8f) * powf(p00, 0.2f);
    Cout[(size_t)q0 * T_ROWS + t1] = powf(d01, 0.8f) * powf(p01, 0.2f);
    Cout[(size_t)q1 * T_ROWS + t0] = powf(d10, 0.8f) * powf(p10, 0.2f);
    Cout[(size_t)q1 * T_ROWS + t1] = powf(d11, 0.8f) * powf(p11, 0.2f);
  }
}

// ---------------- pass 4: Hungarian (maximize) per image, one wave32 per block ----
__device__ __forceinline__ double shfl_down_d(double x, int off) {
  union { double d; int i[2]; } u; u.d = x;
  u.i[0] = __shfl_down(u.i[0], off, 32);
  u.i[1] = __shfl_down(u.i[1], off, 32);
  return u.d;
}
__device__ __forceinline__ double shfl0_d(double x) {
  union { double d; int i[2]; } u; u.d = x;
  u.i[0] = __shfl(u.i[0], 0, 32);
  u.i[1] = __shfl(u.i[1], 0, 32);
  return u.d;
}

__global__ void __launch_bounds__(32) hungarian_kernel(
    const float* __restrict__ Cmat, int* __restrict__ outRows, int* __restrict__ outCols) {
  __shared__ float cst[SQ * SQ];
  __shared__ double u[SQ + 1], vp[SQ + 1], minv[SQ + 1];
  __shared__ int p[SQ + 1], way[SQ + 1];
  __shared__ unsigned char used[SQ + 1];
  const int b = blockIdx.x;
  const int lane = threadIdx.x;
  const double DINF = 1e18;

  for (int idx = lane; idx < SQ * SQ; idx += 32) {
    int i = idx / SQ, j = idx % SQ;
    float v = 0.0f;
    if (j < PERIMG) v = -Cmat[(size_t)(b * 100 + i) * T_ROWS + b * PERIMG + j];  // maximize
    cst[idx] = v;
  }
  for (int j = lane; j <= SQ; j += 32) { u[j] = 0.0; vp[j] = 0.0; p[j] = 0; way[j] = 0; }
  __syncthreads();

  for (int i = 1; i <= SQ; ++i) {
    if (lane == 0) p[0] = i;
    for (int j = lane; j <= SQ; j += 32) { minv[j] = DINF; used[j] = 0; }
    __syncthreads();
    int j0 = 0;
    while (true) {
      if (lane == 0) used[j0] = 1;
      __syncthreads();
      const int i0 = p[j0];
      const double ui0 = u[i0];
      double bestv = DINF;
      int bestj = SQ + 1;
      for (int j = 1 + lane; j <= SQ; j += 32) {
        if (!used[j]) {
          double cur = (double)cst[(i0 - 1) * SQ + (j - 1)] - ui0 - vp[j];
          if (cur < minv[j]) { minv[j] = cur; way[j] = j0; }
          double mv = minv[j];
          if (mv < bestv) { bestv = mv; bestj = j; }  // strict < keeps smallest j per lane
        }
      }
      __syncthreads();
      for (int off = 16; off > 0; off >>= 1) {
        double ov = shfl_down_d(bestv, off);
        int oj = __shfl_down(bestj, off, 32);
        if (ov < bestv || (ov == bestv && oj < bestj)) { bestv = ov; bestj = oj; }
      }
      bestj = __shfl(bestj, 0, 32);
      double delta = shfl0_d(bestv);
      for (int j = lane; j <= SQ; j += 32) {
        if (used[j]) { u[p[j]] += delta; vp[j] -= delta; }
        else         { minv[j] -= delta; }
      }
      __syncthreads();
      j0 = bestj;
      if (p[j0] == 0) break;
    }
    if (lane == 0) {
      int jj = j0;
      while (jj) { int j1 = way[jj]; p[jj] = p[j1]; jj = j1; }
    }
    __syncthreads();
  }

  if (lane == 0) {
    int rr[PERIMG], cc[PERIMG], cnt = 0;
    for (int j = 1; j <= SQ && cnt < PERIMG; ++j) {
      int i = p[j];
      if (i >= 1 && (i - 1) < 100 && (j - 1) < PERIMG) { rr[cnt] = i - 1; cc[cnt] = j - 1; ++cnt; }
    }
    for (int a = 1; a < cnt; ++a) {  // sort by row (matches scipy ordering)
      int kr = rr[a], kc = cc[a], bp = a - 1;
      while (bp >= 0 && rr[bp] > kr) { rr[bp + 1] = rr[bp]; cc[bp + 1] = cc[bp]; --bp; }
      rr[bp + 1] = kr; cc[bp + 1] = kc;
    }
    for (int a = 0; a < cnt; ++a) {
      outRows[b * PERIMG + a] = rr[a];
      outCols[b * PERIMG + a] = cc[a];
    }
  }
}

extern "C" void kernel_launch(void* const* d_in, const int* in_sizes, int n_in,
                              void* d_out, int out_size, void* d_ws, size_t ws_size,
                              hipStream_t stream) {
  const float* pred_masks  = (const float*)d_in[0];  // (16,100,128,128) f32
  const float* pred_logits = (const float*)d_in[1];  // (16,100,80) f32
  const float* tgt_masks   = (const float*)d_in[2];  // (320,256,256) f32
  const int*   tgt_ids     = (const int*)d_in[3];    // (320,) i32

  char* ws = (char*)d_ws;
  _Float16* pm  = (_Float16*)ws;                                       // 52,428,800 B
  _Float16* tmv = (_Float16*)(ws + (size_t)BN_ROWS * HW * 2);          // 10,485,760 B
  float* pmsq = (float*)(ws + (size_t)BN_ROWS * HW * 2 + (size_t)T_ROWS * HW * 2);
  float* tmsq = pmsq + BN_ROWS;

  float* Cout = (float*)d_out;                              // 512000 f32
  int* outRows = (int*)((float*)d_out + 512000);            // 320 i32 (bit pattern)
  int* outCols = (int*)((float*)d_out + 512000 + 320);      // 320 i32

  hipLaunchKernelGGL(pm_sigmoid_kernel, dim3(BN_ROWS), dim3(256), 0, stream,
                     pred_masks, pm, pmsq);
  hipLaunchKernelGGL(tm_resize_kernel, dim3(T_ROWS), dim3(256), 0, stream,
                     tgt_masks, tmv, tmsq);
  hipLaunchKernelGGL(gemm_dice_kernel, dim3(125), dim3(128), 0, stream,
                     pm, tmv, pmsq, tmsq, pred_logits, tgt_ids, Cout);
  hipLaunchKernelGGL(hungarian_kernel, dim3(NB), dim3(32), 0, stream,
                     Cout, outRows, outCols);
}
